// LSTMModel_55765855371539
// MI455X (gfx1250) — matrix operand order repaired and checked by reading, not compiled, not run on hardware
//
#include <hip/hip_runtime.h>
#include <stdint.h>

#define H 128
#define T_ENC 64
#define T_DEC 32
#define B_TILE 32
#define ACT_STRIDE 264   // halves per row: 128 (x) + 128 (h) + 8 pad (bank-conflict break)

// LDS partition (halves): cell0 weights 8 waves * 4 gates * 8 slices * 512 = 131072
#define W0_HALVES   131072
#define ACT_HALVES  (B_TILE * ACT_STRIDE)      // 8448
#define SMEM_BYTES  (W0_HALVES * 2 + 2 * ACT_HALVES * 2 + B_TILE * H * 4)  // 312320 B <= 320KB

typedef __attribute__((ext_vector_type(16))) _Float16 v16h;
typedef __attribute__((ext_vector_type(8)))  _Float16 v8h;
typedef __attribute__((ext_vector_type(8)))  float    v8f;
typedef __attribute__((ext_vector_type(4)))  float    v4f;

#if __has_builtin(__builtin_amdgcn_exp2f)
#define EXP2F(x) __builtin_amdgcn_exp2f(x)
#else
#define EXP2F(x) exp2f(x)
#endif
#if __has_builtin(__builtin_amdgcn_rcpf)
#define RCPF(x) __builtin_amdgcn_rcpf(x)
#else
#define RCPF(x) (1.0f/(x))
#endif

__device__ __forceinline__ float fast_exp(float x)  { return EXP2F(x * 1.4426950408889634f); }
__device__ __forceinline__ float fast_sig(float x)  { return RCPF(1.0f + fast_exp(-x)); }
__device__ __forceinline__ float fast_tanh(float x) { float e = fast_exp(-2.0f * x);
                                                      return (1.0f - e) * RCPF(1.0f + e); }

// Load one WMMA B-tile (32x16 f16) from a row-major f32 weight matrix W[4H][H].
// Lane (n, hi): needs W[colBase+n][kBase + hi*16 + j], j=0..15  (16 consecutive floats).
__device__ __forceinline__ v16h load_wtile(const float* __restrict__ p /* = W + row*H + col */) {
  const v4f* q = (const v4f*)p;
  v4f f0 = q[0], f1 = q[1], f2 = q[2], f3 = q[3];
  v16h r;
  r[0]=(_Float16)f0.x; r[1]=(_Float16)f0.y; r[2]=(_Float16)f0.z; r[3]=(_Float16)f0.w;
  r[4]=(_Float16)f1.x; r[5]=(_Float16)f1.y; r[6]=(_Float16)f1.z; r[7]=(_Float16)f1.w;
  r[8]=(_Float16)f2.x; r[9]=(_Float16)f2.y; r[10]=(_Float16)f2.z; r[11]=(_Float16)f2.w;
  r[12]=(_Float16)f3.x; r[13]=(_Float16)f3.y; r[14]=(_Float16)f3.z; r[15]=(_Float16)f3.w;
  return r;
}

// Load one WMMA A-tile (16x32 f16) from LDS activations (row stride ACT_STRIDE halves).
__device__ __forceinline__ v16h load_atile(const _Float16* act, int rowBase, int kOff,
                                           int laneN, int laneHi) {
  const _Float16* p = act + (rowBase + laneN) * ACT_STRIDE + kOff + laneHi * 16;
  v8h lo = *(const v8h*)p;
  v8h hi = *(const v8h*)(p + 8);
  v16h r;
#pragma unroll
  for (int j = 0; j < 8; ++j) { r[j] = lo[j]; r[j + 8] = hi[j]; }
  return r;
}

// Cell0 B-tiles live in LDS, stored in the exact per-lane register image:
// tile tIdx: halves [tIdx*512 + lane*8] (j=0..7) and [tIdx*512 + 256 + lane*8] (j=8..15).
__device__ __forceinline__ v16h load_btile(const _Float16* wlds, int tIdx, int lane) {
  const _Float16* p = wlds + tIdx * 512 + lane * 8;
  v8h lo = *(const v8h*)p;
  v8h hi = *(const v8h*)(p + 256);
  v16h r;
#pragma unroll
  for (int j = 0; j < 8; ++j) { r[j] = lo[j]; r[j + 8] = hi[j]; }
  return r;
}

__device__ __forceinline__ void store_btile(_Float16* wlds, int tIdx, int lane, v16h t) {
  v8h lo, hi;
#pragma unroll
  for (int j = 0; j < 8; ++j) { lo[j] = t[j]; hi[j] = t[j + 8]; }
  _Float16* p = wlds + tIdx * 512 + lane * 8;
  *(v8h*)p = lo;
  *(v8h*)(p + 256) = hi;
}

// Shared elementwise LSTM gate update; h packed in registers (LDS store after barrier).
__device__ __forceinline__ void gates_ew(v8f (&acc)[4][2], v8f (&cst)[2],
                                         float bi, float bf, float bg, float bo,
                                         v8h (&hh)[2]) {
#pragma unroll
  for (int rt = 0; rt < 2; ++rt)
#pragma unroll
    for (int e = 0; e < 8; ++e) {
      float gi = fast_sig (acc[0][rt][e] + bi);
      float gf = fast_sig (acc[1][rt][e] + bf);
      float gg = fast_tanh(acc[2][rt][e] + bg);
      float go = fast_sig (acc[3][rt][e] + bo);
      float c  = gf * cst[rt][e] + gi * gg;
      cst[rt][e] = c;
      hh[rt][e] = (_Float16)(go * fast_tanh(c));
    }
}

// Cell 0: B-tiles streamed from LDS. S0=4 skips the zero-x K half in decode.
template<int S0>
__device__ __forceinline__ void cell0_forward(const _Float16* act, const _Float16* wlds,
                                              int w, v8f (&cst)[2],
                                              float bi, float bf, float bg, float bo,
                                              v8h (&hh)[2], int lane, int laneN, int laneHi) {
  const v8f zz = {0.f,0.f,0.f,0.f,0.f,0.f,0.f,0.f};
  v8f acc[4][2] = {{zz,zz},{zz,zz},{zz,zz},{zz,zz}};
#pragma unroll
  for (int s = S0; s < 8; ++s) {
    v16h a0 = load_atile(act, 0,  s * 32, laneN, laneHi);
    v16h a1 = load_atile(act, 16, s * 32, laneN, laneHi);
#pragma unroll
    for (int g = 0; g < 4; ++g) {
      v16h b = load_btile(wlds, (w * 4 + g) * 8 + s, lane);
      acc[g][0] = __builtin_amdgcn_wmma_f32_16x16x32_f16(false, a0, false, b, (short)0, acc[g][0], false, false);
      acc[g][1] = __builtin_amdgcn_wmma_f32_16x16x32_f16(false, a1, false, b, (short)0, acc[g][1], false, false);
    }
  }
  gates_ew(acc, cst, bi, bf, bg, bo, hh);
}

// Cell 1: B-tiles resident in VGPRs.
__device__ __forceinline__ void cell1_forward(const _Float16* act, const v16h (&wt)[4][8],
                                              v8f (&cst)[2],
                                              float bi, float bf, float bg, float bo,
                                              v8h (&hh)[2], int laneN, int laneHi) {
  const v8f zz = {0.f,0.f,0.f,0.f,0.f,0.f,0.f,0.f};
  v8f acc[4][2] = {{zz,zz},{zz,zz},{zz,zz},{zz,zz}};
#pragma unroll
  for (int s = 0; s < 8; ++s) {
    v16h a0 = load_atile(act, 0,  s * 32, laneN, laneHi);
    v16h a1 = load_atile(act, 16, s * 32, laneN, laneHi);
#pragma unroll
    for (int g = 0; g < 4; ++g) {
      acc[g][0] = __builtin_amdgcn_wmma_f32_16x16x32_f16(false, a0, false, wt[g][s], (short)0, acc[g][0], false, false);
      acc[g][1] = __builtin_amdgcn_wmma_f32_16x16x32_f16(false, a1, false, wt[g][s], (short)0, acc[g][1], false, false);
    }
  }
  gates_ew(acc, cst, bi, bf, bg, bo, hh);
}

// Store this wave's h slice (held in registers across the barrier) into LDS buffers.
__device__ __forceinline__ void store_h(const v8h (&hh)[2],
                                        _Float16* dst0, int col0,
                                        _Float16* dst1, int col1,
                                        int laneN, int laneHi) {
#pragma unroll
  for (int rt = 0; rt < 2; ++rt)
#pragma unroll
    for (int e = 0; e < 8; ++e) {
      int row = rt * 16 + laneHi * 8 + e;               // C/D layout: vgpr e, hi-half = +8
      dst0[row * ACT_STRIDE + col0 + laneN] = hh[rt][e];
      if (dst1) dst1[row * ACT_STRIDE + col1 + laneN] = hh[rt][e];
    }
}

// Async (CDNA5) prefetch of x_t for this wave's 4 rows into the f32 LDS stage.
__device__ __forceinline__ void async_load_x(const float* __restrict__ x, int rowBlock, int t,
                                             int w, int lane, float* xstage) {
  int row = w * 4 + (lane >> 3);
  int cb  = (lane & 7) * 16;
  const float* src = x + ((size_t)(rowBlock + row) * T_ENC + t) * H + cb;
  unsigned lds = (unsigned)(uintptr_t)&xstage[row * H + cb];
#pragma unroll
  for (int j = 0; j < 4; ++j) {
    uint64_t ga = (uint64_t)(uintptr_t)src + (uint64_t)(j * 16);
    unsigned  la = lds + j * 16;
    asm volatile("global_load_async_to_lds_b128 %0, %1, off"
                 :: "v"(la), "v"(ga) : "memory");
  }
}

__global__ __launch_bounds__(256) void lstm_persist_kernel(
    const float* __restrict__ x,
    const float* __restrict__ Wih0, const float* __restrict__ Whh0, const float* __restrict__ bias0,
    const float* __restrict__ Wih1, const float* __restrict__ Whh1, const float* __restrict__ bias1,
    const float* __restrict__ fcW,  const float* __restrict__ fcb,
    float* __restrict__ out) {
  extern __shared__ __align__(16) char smem[];
  _Float16* w0lds  = (_Float16*)smem;                    // 256KB: cell0 weight tiles
  _Float16* actA   = w0lds + W0_HALVES;                  // [x_t | h0]
  _Float16* actB   = actA + ACT_HALVES;                  // [h0' | h1]
  float*    xstage = (float*)(actB + ACT_HALVES);        // raw f32 x_t (async landing zone)

  const int tid    = threadIdx.x;
  const int w      = tid >> 5;        // wave: owns hidden columns [w*16, w*16+16)
  const int lane   = tid & 31;
  const int laneN  = lane & 15;
  const int laneHi = lane >> 4;
  const int rowBlock = blockIdx.x * B_TILE;

  // ---- cell1 weights resident in VGPRs (4 gates x 8 K-slices = 256 VGPRs) ----
  // ---- cell0 weights staged into LDS in B-register image ----
  v16h wt1[4][8];
#pragma unroll
  for (int g = 0; g < 4; ++g) {
    int row = g * H + w * 16 + laneN;       // gate output index
#pragma unroll
    for (int s = 0; s < 8; ++s) {
      int col = (s & 3) * 32 + laneHi * 16; // K within Wih (s<4) or Whh (s>=4)
      wt1[g][s] = load_wtile(((s < 4) ? Wih1 : Whh1) + row * H + col);
      v16h t0  = load_wtile(((s < 4) ? Wih0 : Whh0) + row * H + col);
      store_btile(w0lds, (w * 4 + g) * 8 + s, lane, t0);
    }
  }
  const int bcol = w * 16 + laneN;
  float bi0 = bias0[0*H + bcol], bf0 = bias0[1*H + bcol], bg0 = bias0[2*H + bcol], bo0 = bias0[3*H + bcol];
  float bi1 = bias1[0*H + bcol], bf1 = bias1[1*H + bcol], bg1 = bias1[2*H + bcol], bo1 = bias1[3*H + bcol];

  // zero-init activation buffers + c state
  for (int i = tid; i < ACT_HALVES; i += 256) {
    actA[i] = (_Float16)0.f; actB[i] = (_Float16)0.f;
  }
  const v8f zz = {0.f,0.f,0.f,0.f,0.f,0.f,0.f,0.f};
  v8f c0[2] = {zz, zz}, c1[2] = {zz, zz};
  v8h hh0[2], hh1[2];

  async_load_x(x, rowBlock, 0, w, lane, xstage);   // prefetch x_0
  __syncthreads();

  for (int t = 0; t < T_ENC + T_DEC; ++t) {
    const bool enc = (t < T_ENC);
    if (enc) {
      asm volatile("s_wait_asynccnt 0x0" ::: "memory");
      // convert staged f32 -> f16 into actA[:, 0:128]; each wave converts its own rows
      int row = tid >> 3;
      int cb  = (tid & 7) * 16;
#pragma unroll
      for (int j = 0; j < 16; ++j)
        actA[row * ACT_STRIDE + cb + j] = (_Float16)xstage[row * H + cb + j];
    }
    __syncthreads();                                    // x ready; prev-step h writes ready
    if (enc && (t + 1) < T_ENC) async_load_x(x, rowBlock, t + 1, w, lane, xstage);

    // ---- cell 0: A = [x_t | h0], B from LDS ----
    if (enc) cell0_forward<0>(actA, w0lds, w, c0, bi0, bf0, bg0, bo0, hh0, lane, laneN, laneHi);
    else     cell0_forward<4>(actA, w0lds, w, c0, bi0, bf0, bg0, bo0, hh0, lane, laneN, laneHi);
    __syncthreads();                                    // everyone done reading actA
    store_h(hh0, actB, /*h0' -> */ w * 16,
                 actA, /*h0  -> */ H + w * 16, laneN, laneHi);
    __syncthreads();                                    // actB[:,0:128] = h0' ready

    // ---- cell 1: A = [h0' | h1], B in VGPRs ----
    cell1_forward(actB, wt1, c1, bi1, bf1, bg1, bo1, hh1, laneN, laneHi);
    __syncthreads();                                    // everyone done reading actB
    store_h(hh1, actB, /*h1 -> */ H + w * 16,
                 (_Float16*)nullptr, 0, laneN, laneHi);
  }
  __syncthreads();                                      // final h1 in actB[:,128:256]

  // ---- FC: out_row = h1 @ fcW.T + fcb (this wave's 16 output columns) ----
  v16h fwt[4];
#pragma unroll
  for (int s = 0; s < 4; ++s)
    fwt[s] = load_wtile(fcW + (w * 16 + laneN) * H + s * 32 + laneHi * 16);
  float fb = fcb[w * 16 + laneN];

#pragma unroll
  for (int rt = 0; rt < 2; ++rt) {
    v8f facc = zz;
#pragma unroll
    for (int s = 0; s < 4; ++s) {
      v16h a = load_atile(actB, rt * 16, H + s * 32, laneN, laneHi);
      facc = __builtin_amdgcn_wmma_f32_16x16x32_f16(false, a, false, fwt[s],
                                                    (short)0, facc, false, false);
    }
#pragma unroll
    for (int e = 0; e < 8; ++e) {
      int row = rowBlock + rt * 16 + laneHi * 8 + e;
      out[((size_t)row * (T_ENC + 1) + T_ENC) * H + w * 16 + laneN] = facc[e] + fb;
    }
  }
}

// out[:, 0:64, :] = x  (pure bandwidth, 16B granularity, non-temporal)
__global__ __launch_bounds__(256) void copy_x_kernel(const v4f* __restrict__ x4,
                                                     v4f* __restrict__ out4) {
  int i = blockIdx.x * 256 + threadIdx.x;   // 0 .. B*T*H/4 - 1
  int b = i >> 11;                          // 64*128/4 = 2048 vec4 per batch row
  int r = i & 2047;
  v4f v = __builtin_nontemporal_load(&x4[i]);
  __builtin_nontemporal_store(v, &out4[(size_t)b * 2080 + r]);   // 65*128/4 = 2080
}

extern "C" void kernel_launch(void* const* d_in, const int* in_sizes, int n_in,
                              void* d_out, int out_size, void* d_ws, size_t ws_size,
                              hipStream_t stream) {
  const float* x    = (const float*)d_in[0];
  const float* Wih0 = (const float*)d_in[1];
  const float* Whh0 = (const float*)d_in[2];
  const float* b0   = (const float*)d_in[3];
  const float* Wih1 = (const float*)d_in[4];
  const float* Whh1 = (const float*)d_in[5];
  const float* b1   = (const float*)d_in[6];
  const float* fcW  = (const float*)d_in[7];
  const float* fcb  = (const float*)d_in[8];
  float* out = (float*)d_out;

  // 4096*64*128 / 4 / 256 = 32768 blocks
  copy_x_kernel<<<32768, 256, 0, stream>>>((const v4f*)x, (v4f*)out);
  // 4096 / 32 rows per workgroup = 128 persistent workgroups, 8 waves each.
  // 305KB dynamic LDS: 256KB cell0 weights + 2 activation buffers + x stage.
  lstm_persist_kernel<<<128, 256, SMEM_BYTES, stream>>>(x, Wih0, Whh0, b0, Wih1, Whh1, b1,
                                                        fcW, fcb, out);
}
